// RWKV6Wrapper_47476568490663
// MI455X (gfx1250) — matrix operand order repaired and checked by reading, not compiled, not run on hardware
//
#include <hip/hip_runtime.h>
#include <hip/hip_bf16.h>

// RWKV6 forward for gfx1250 (MI455X). wave32, WMMA 16x16x32 f16.

typedef __attribute__((ext_vector_type(16))) _Float16 v16h;
typedef __attribute__((ext_vector_type(8)))  _Float16 v8h;
typedef __attribute__((ext_vector_type(8)))  float    v8f;
typedef __attribute__((ext_vector_type(4)))  float    v4f;

#define RW_B  2
#define RW_T  512
#define RW_D  768
#define RW_H  12
#define RW_N  64
#define RW_F  2688
#define RW_ROWS (RW_B * RW_T)   // 1024

// ---------------------------------------------------------------------------
// Epilogue ops for the fused GEMM
// 0 = none, 1 = silu, 2 = relu^2, 3 = sigmoid
__device__ __forceinline__ float rw_epi(float x, int epi) {
  if (epi == 1) return x / (1.0f + __expf(-x));
  if (epi == 2) { float t = fmaxf(x, 0.0f); return t * t; }
  if (epi == 3) return 1.0f / (1.0f + __expf(-x));
  return x;
}

__device__ __forceinline__ v16h rw_frag(const v8h* p) {
  v8h lo = p[0], hi = p[1];
  return __builtin_shufflevector(lo, hi,
      0,1,2,3,4,5,6,7,8,9,10,11,12,13,14,15);
}

// ---------------------------------------------------------------------------
// Fused GEMM: out[M,Nn] = epi(A[M,K] @ W[K,Nn]) * gate + residual
// Block = 256 threads = 8 waves. Block tile 128(M) x 64(N), K-step 32.
// Wave (wm 0..3, wn 0..1) computes 32x32 via four v_wmma_f32_16x16x32_f16
// (2 A-fragments x 2 B-fragments -> 8 ds_load_b128 : 4 WMMA per K-step).
//
// LDS holds tiles in *fragment-major* f16 layout so that:
//   - cooperative stores are single ds_store_b128 per chunk,
//   - per-lane fragment loads are two aligned ds_load_b128, zero shuffles.
// ISA lane maps used:
//   A (16x32 f16): lane = M + 16*((K>>3)&1), elem = (K&7) + 8*(K>>4)
//   B (32x16 f16): lane = N + 16*((K>>3)&1), elem = (K&7) + 8*(K>>4)
// Global loads for tile k+1 are register-staged before tile k's WMMAs.
__global__ __launch_bounds__(256)
void rw_gemm_wmma(const float* __restrict__ A, const float* __restrict__ W,
                  float* __restrict__ outp, int M, int K, int Nn,
                  int epi, const float* __restrict__ gate,
                  const float* __restrict__ residual)
{
  // [tile][lane][half-of-fragment]
  __shared__ v8h As_f[8][32][2];   // 8 row-tiles of 16 : 8 KB
  __shared__ v8h Bs_f[4][32][2];   // 4 col-tiles of 16 : 4 KB

  const int tid  = threadIdx.x;
  const int lane = tid & 31;
  const int wave = tid >> 5;     // 0..7
  const int wm   = wave & 3;     // 32-row slice (two 16-row tiles)
  const int wn   = wave >> 2;    // 32-col slice (two 16-col tiles)
  const int row0 = blockIdx.y * 128;
  const int col0 = blockIdx.x * 64;

  // A loader: row ar (0..127), two K-chunks of 8 starting at aq0 in {0,2}
  const int ar    = tid >> 1;
  const int aq0   = (tid & 1) * 2;
  const int aTile = ar >> 4;
  const int aLane = ar & 15;
  const int aHalf = tid & 1;           // aq0>>1
  // B loader: col bc (0..63), K-chunk bq (0..3) of 8
  const int bc    = tid & 63;
  const int bq    = tid >> 6;
  const int bTile = bc >> 4;
  const int bLane = (bc & 15) + 16 * (bq & 1);
  const int bHalf = bq >> 1;

  v8f c00 = {}, c01 = {}, c10 = {}, c11 = {};

  // register staging for the next K tile
  v4f sa[4];
  float sb[8];

  auto stage = [&](int kbase) {
    const float* ap = A + (size_t)(row0 + ar) * K + kbase + aq0 * 8;
    sa[0] = *(const v4f*)ap;             // 16 contiguous floats:
    sa[1] = *(const v4f*)(ap + 4);       // 4x global_load_b128
    sa[2] = *(const v4f*)(ap + 8);
    sa[3] = *(const v4f*)(ap + 12);
    const float* wp = W + (size_t)(kbase + bq * 8) * Nn + col0 + bc;
    #pragma unroll
    for (int j = 0; j < 8; ++j) sb[j] = wp[(size_t)j * Nn];  // lane-coalesced
  };

  auto commit = [&]() {
    v8h pa0, pa1;
    #pragma unroll
    for (int j = 0; j < 4; ++j) {
      pa0[j]     = (_Float16)sa[0][j];
      pa0[4 + j] = (_Float16)sa[1][j];
      pa1[j]     = (_Float16)sa[2][j];
      pa1[4 + j] = (_Float16)sa[3][j];
    }
    As_f[aTile][aLane][aHalf]      = pa0;    // chunk aq0   -> lane M
    As_f[aTile][aLane + 16][aHalf] = pa1;    // chunk aq0+1 -> lane M+16
    v8h pb;
    #pragma unroll
    for (int j = 0; j < 8; ++j) pb[j] = (_Float16)sb[j];
    Bs_f[bTile][bLane][bHalf] = pb;          // ds_store_b128
  };

  stage(0);
  for (int kk = 0; kk < K; kk += 32) {
    commit();
    __syncthreads();
    if (kk + 32 < K) stage(kk + 32);         // overlaps with WMMA below

    v16h a0 = rw_frag(As_f[wm * 2 + 0][lane]);
    v16h a1 = rw_frag(As_f[wm * 2 + 1][lane]);
    v16h b0 = rw_frag(Bs_f[wn * 2 + 0][lane]);
    v16h b1 = rw_frag(Bs_f[wn * 2 + 1][lane]);

    c00 = __builtin_amdgcn_wmma_f32_16x16x32_f16(false, a0, false, b0,
                                                 (short)0, c00, false, false);
    c01 = __builtin_amdgcn_wmma_f32_16x16x32_f16(false, a0, false, b1,
                                                 (short)0, c01, false, false);
    c10 = __builtin_amdgcn_wmma_f32_16x16x32_f16(false, a1, false, b0,
                                                 (short)0, c10, false, false);
    c11 = __builtin_amdgcn_wmma_f32_16x16x32_f16(false, a1, false, b1,
                                                 (short)0, c11, false, false);
    __syncthreads();
  }

  // C/D layout: element r of v8f is C[M = (lane>>4)*8 + r][N = lane&15].
  const int hl = lane & 15;
  const int kg = lane >> 4;
  #pragma unroll
  for (int mi = 0; mi < 2; ++mi) {
    #pragma unroll
    for (int ni = 0; ni < 2; ++ni) {
      const v8f& c = (mi == 0) ? (ni == 0 ? c00 : c01)
                               : (ni == 0 ? c10 : c11);
      #pragma unroll
      for (int r = 0; r < 8; ++r) {
        int grow = row0 + (wm * 2 + mi) * 16 + kg * 8 + r;
        int gcol = col0 + (wn * 2 + ni) * 16 + hl;
        size_t idx = (size_t)grow * Nn + gcol;
        float v = rw_epi(c[r], epi);
        if (gate)     v *= gate[idx];
        if (residual) v += residual[idx];
        outp[idx] = v;
      }
    }
  }
}

// ---------------------------------------------------------------------------
// LayerNorm over D=768; one block per row.
__global__ __launch_bounds__(256)
void rw_ln(const float* __restrict__ x, const float* __restrict__ w,
           const float* __restrict__ b, float* __restrict__ y)
{
  __shared__ float xb[RW_D];
  __shared__ float red[256];
  const int row = blockIdx.x;
  const int tid = threadIdx.x;
  const float* xr = x + (size_t)row * RW_D;

  float s = 0.0f;
  for (int d = tid; d < RW_D; d += 256) { float v = xr[d]; xb[d] = v; s += v; }
  red[tid] = s; __syncthreads();
  for (int o = 128; o > 0; o >>= 1) {
    if (tid < o) red[tid] += red[tid + o];
    __syncthreads();
  }
  float mean = red[0] * (1.0f / RW_D);
  __syncthreads();
  float s2 = 0.0f;
  for (int d = tid; d < RW_D; d += 256) { float t = xb[d] - mean; s2 += t * t; }
  red[tid] = s2; __syncthreads();
  for (int o = 128; o > 0; o >>= 1) {
    if (tid < o) red[tid] += red[tid + o];
    __syncthreads();
  }
  float inv = rsqrtf(red[0] * (1.0f / RW_D) + 1e-5f);
  for (int d = tid; d < RW_D; d += 256)
    y[(size_t)row * RW_D + d] = (xb[d] - mean) * inv * w[d] + b[d];
}

// ln0 applied on transposed input: x is [B,D,T]; write h [B*T, D].
__global__ __launch_bounds__(256)
void rw_ln0_transpose(const float* __restrict__ x, const float* __restrict__ w,
                      const float* __restrict__ b, float* __restrict__ h)
{
  __shared__ float xb[RW_D];
  __shared__ float red[256];
  const int row = blockIdx.x;            // b*T + t
  const int bb = row / RW_T, t = row % RW_T;
  const int tid = threadIdx.x;

  float s = 0.0f;
  for (int d = tid; d < RW_D; d += 256) {
    float v = x[((size_t)bb * RW_D + d) * RW_T + t];
    xb[d] = v; s += v;
  }
  red[tid] = s; __syncthreads();
  for (int o = 128; o > 0; o >>= 1) {
    if (tid < o) red[tid] += red[tid + o];
    __syncthreads();
  }
  float mean = red[0] * (1.0f / RW_D);
  __syncthreads();
  float s2 = 0.0f;
  for (int d = tid; d < RW_D; d += 256) { float t2 = xb[d] - mean; s2 += t2 * t2; }
  red[tid] = s2; __syncthreads();
  for (int o = 128; o > 0; o >>= 1) {
    if (tid < o) red[tid] += red[tid + o];
    __syncthreads();
  }
  float inv = rsqrtf(red[0] * (1.0f / RW_D) + 1e-5f);
  for (int d = tid; d < RW_D; d += 256)
    h[(size_t)row * RW_D + d] = (xb[d] - mean) * inv * w[d] + b[d];
}

// h [B*T,D] -> out [B,D,T]
__global__ void rw_transpose_out(const float* __restrict__ h, float* __restrict__ outp)
{
  int idx = blockIdx.x * 256 + threadIdx.x;
  if (idx >= RW_B * RW_D * RW_T) return;
  int t = idx % RW_T;
  int d = (idx / RW_T) % RW_D;
  int b = idx / (RW_T * RW_D);
  outp[idx] = h[((size_t)b * RW_T + t) * RW_D + d];
}

// ---------------------------------------------------------------------------
// Token-shift + LoRA mixing for tmix: produces xw/xk/xv/xr/xg and decay w.
__global__ __launch_bounds__(256)
void rw_tmix_mix(const float* __restrict__ xln,
                 const float* __restrict__ maa_x, const float* __restrict__ maa5,
                 const float* __restrict__ tm_w1, const float* __restrict__ tm_w2,
                 const float* __restrict__ td_w1, const float* __restrict__ td_w2,
                 const float* __restrict__ decay,
                 float* __restrict__ pxw, float* __restrict__ pxk,
                 float* __restrict__ pxv, float* __restrict__ pxr,
                 float* __restrict__ pxg, float* __restrict__ wdec)
{
  __shared__ float tm[RW_D];     // (x + sx*maa_x), later reused as xw vector
  __shared__ float mix[160];     // 5*LM
  __shared__ float dw[64];       // LD
  const int row = blockIdx.x;
  const int t = row % RW_T;
  const int tid = threadIdx.x;
  const float* xr0 = xln + (size_t)row * RW_D;
  const float* xp  = (t > 0) ? (xr0 - RW_D) : nullptr;

  for (int d = tid; d < RW_D; d += 256) {
    float xv0 = xr0[d];
    float sx  = (xp ? xp[d] : 0.0f) - xv0;
    tm[d] = xv0 + sx * maa_x[d];
  }
  __syncthreads();

  if (tid < 160) {
    float acc = 0.0f;
    for (int k = 0; k < RW_D; ++k) acc += tm[k] * tm_w1[k * 160 + tid];
    mix[tid] = tanhf(acc);
  }
  __syncthreads();

  for (int d = tid; d < RW_D; d += 256) {
    float xv0 = xr0[d];
    float sx  = (xp ? xp[d] : 0.0f) - xv0;
    float m[5];
    #pragma unroll
    for (int f = 0; f < 5; ++f) {
      float acc = 0.0f;
      #pragma unroll
      for (int k = 0; k < 32; ++k)
        acc += mix[f * 32 + k] * tm_w2[(size_t)(f * 32 + k) * RW_D + d];
      m[f] = acc;
    }
    size_t o = (size_t)row * RW_D + d;
    float xw = xv0 + sx * (maa5[0 * RW_D + d] + m[0]);
    pxw[o] = xw;
    pxk[o] = xv0 + sx * (maa5[1 * RW_D + d] + m[1]);
    pxv[o] = xv0 + sx * (maa5[2 * RW_D + d] + m[2]);
    pxr[o] = xv0 + sx * (maa5[3 * RW_D + d] + m[3]);
    pxg[o] = xv0 + sx * (maa5[4 * RW_D + d] + m[4]);
    tm[d] = xw;   // reuse LDS for the decay LoRA input
  }
  __syncthreads();

  if (tid < 64) {
    float acc = 0.0f;
    for (int k = 0; k < RW_D; ++k) acc += tm[k] * td_w1[k * 64 + tid];
    dw[tid] = tanhf(acc);
  }
  __syncthreads();

  for (int d = tid; d < RW_D; d += 256) {
    float acc = decay[d];
    #pragma unroll
    for (int j = 0; j < 64; ++j) acc += dw[j] * td_w2[(size_t)j * RW_D + d];
    wdec[(size_t)row * RW_D + d] = __expf(-__expf(acc));
  }
}

// ---------------------------------------------------------------------------
// WKV6 scan: one block per (b,h), 64 threads; thread j holds state column j.
__global__ __launch_bounds__(64)
void rw_wkv(const float* __restrict__ r, const float* __restrict__ k,
            const float* __restrict__ v, const float* __restrict__ w,
            const float* __restrict__ u, float* __restrict__ outp)
{
  const int bh = blockIdx.x;             // b*H + h
  const int b = bh / RW_H, hh = bh % RW_H;
  const int j = threadIdx.x;
  __shared__ float rl[RW_N], kl[RW_N], wl[RW_N], ul[RW_N];

  ul[j] = u[hh * RW_N + j];
  float s[RW_N];
  #pragma unroll
  for (int i = 0; i < RW_N; ++i) s[i] = 0.0f;

  const size_t base = (size_t)b * RW_T * RW_D + (size_t)hh * RW_N;
  __syncthreads();

  for (int t = 0; t < RW_T; ++t) {
    size_t off = base + (size_t)t * RW_D + j;
    float vt = v[off];
    rl[j] = r[off]; kl[j] = k[off]; wl[j] = w[off];
    __syncthreads();
    float acc = 0.0f, accu = 0.0f;
    #pragma unroll
    for (int i = 0; i < RW_N; ++i) {
      float ri = rl[i], ki = kl[i];
      acc  += ri * s[i];
      accu += ri * ul[i] * ki;
      s[i]  = ki * vt + wl[i] * s[i];
    }
    outp[off] = acc + accu * vt;
    __syncthreads();
  }
}

// ---------------------------------------------------------------------------
// GroupNorm (H groups of N) * gate : one block per (row, head), 64 threads.
__global__ __launch_bounds__(64)
void rw_gn_gate(const float* __restrict__ xin, const float* __restrict__ wgt,
                const float* __restrict__ bia, const float* __restrict__ g,
                float* __restrict__ outp)
{
  const int rh = blockIdx.x;
  const int row = rh / RW_H, hh = rh % RW_H;
  const int n = threadIdx.x;
  __shared__ float red[RW_N];
  size_t off = (size_t)row * RW_D + hh * RW_N + n;
  float x = xin[off];
  red[n] = x; __syncthreads();
  for (int o = 32; o > 0; o >>= 1) {
    if (n < o) red[n] += red[n + o];
    __syncthreads();
  }
  float mean = red[0] * (1.0f / RW_N);
  __syncthreads();
  float d0 = x - mean;
  red[n] = d0 * d0; __syncthreads();
  for (int o = 32; o > 0; o >>= 1) {
    if (n < o) red[n] += red[n + o];
    __syncthreads();
  }
  float inv = rsqrtf(red[0] * (1.0f / RW_N) + 64e-5f);
  float xn = d0 * inv;
  outp[off] = (xn * wgt[hh * RW_N + n] + bia[hh * RW_N + n]) * g[off];
}

// ---------------------------------------------------------------------------
// cmix token-shift mixing (elementwise)
__global__ void rw_cmix_mix(const float* __restrict__ xln,
                            const float* __restrict__ mk,
                            const float* __restrict__ mr,
                            float* __restrict__ xk, float* __restrict__ xr)
{
  int idx = blockIdx.x * 256 + threadIdx.x;
  if (idx >= RW_ROWS * RW_D) return;
  int d = idx % RW_D;
  int row = idx / RW_D;
  int t = row % RW_T;
  float x = xln[idx];
  float prev = (t > 0) ? xln[idx - RW_D] : 0.0f;
  float sx = prev - x;
  xk[idx] = x + sx * mk[d];
  xr[idx] = x + sx * mr[d];
}

// ---------------------------------------------------------------------------
extern "C" void kernel_launch(void* const* d_in, const int* in_sizes, int n_in,
                              void* d_out, int out_size, void* d_ws, size_t ws_size,
                              hipStream_t stream)
{
  (void)in_sizes; (void)n_in; (void)out_size; (void)ws_size;

  // Inputs in setup_inputs() dict order:
  const float* x      = (const float*)d_in[0];   // [B,D,T]
  const float* ln0_w  = (const float*)d_in[1];
  const float* ln0_b  = (const float*)d_in[2];
  const float* ln1_w  = (const float*)d_in[3];   // [4,D]
  const float* ln1_b  = (const float*)d_in[4];
  const float* ln2_w  = (const float*)d_in[5];
  const float* ln2_b  = (const float*)d_in[6];
  const float* lnx_w  = (const float*)d_in[7];
  const float* lnx_b  = (const float*)d_in[8];
  const float* maa_x  = (const float*)d_in[9];   // [4,D]
  const float* maa_5  = (const float*)d_in[10];  // [4,5,D]
  const float* tm_w1  = (const float*)d_in[11];  // [4,D,160]
  const float* tm_w2  = (const float*)d_in[12];  // [4,5,32,D]
  const float* td_w1  = (const float*)d_in[13];  // [4,D,64]
  const float* td_w2  = (const float*)d_in[14];  // [4,64,D]
  const float* decay  = (const float*)d_in[15];  // [4,D]
  const float* faaaa  = (const float*)d_in[16];  // [4,H,N]
  const float* Wr     = (const float*)d_in[17];  // [4,D,D]
  const float* Wk     = (const float*)d_in[18];
  const float* Wv     = (const float*)d_in[19];
  const float* Wg     = (const float*)d_in[20];
  const float* Wo     = (const float*)d_in[21];
  const float* cm_k   = (const float*)d_in[22];  // [4,D]
  const float* cm_r   = (const float*)d_in[23];
  const float* Wkf    = (const float*)d_in[24];  // [4,D,F]
  const float* Wvf    = (const float*)d_in[25];  // [4,F,D]
  const float* Wrf    = (const float*)d_in[26];  // [4,D,D]

  const size_t DSZ = (size_t)RW_ROWS * RW_D;     // 1024*768
  float* ws = (float*)d_ws;
  float* h     = ws;                // persistent hidden state [B*T, D]
  float* xln   = h    + DSZ;
  float* bxw   = xln  + DSZ;
  float* bxk   = bxw  + DSZ;
  float* bxv   = bxk  + DSZ;
  float* bxr   = bxv  + DSZ;
  float* bxg   = bxr  + DSZ;
  float* wdec  = bxg  + DSZ;
  float* rr    = wdec + DSZ;        // also reused as sigmoid(r) in cmix
  float* kkb   = rr   + DSZ;
  float* vvb   = kkb  + DSZ;
  float* ggb   = vvb  + DSZ;
  float* wkvo  = ggb  + DSZ;
  float* ag    = wkvo + DSZ;
  float* kf    = ag   + DSZ;        // [B*T, F]

  const dim3 blk256(256);
  const dim3 gemmDD(RW_D / 64, RW_ROWS / 128);   // (12, 8)
  const dim3 gemmDF(RW_F / 64, RW_ROWS / 128);   // (42, 8)

  // h = ln0(transpose(x))
  rw_ln0_transpose<<<RW_ROWS, blk256, 0, stream>>>(x, ln0_w, ln0_b, h);

  for (int i = 0; i < 4; ++i) {
    const size_t oD  = (size_t)i * RW_D;
    const size_t oDD = (size_t)i * RW_D * RW_D;

    // ---- tmix ----
    rw_ln<<<RW_ROWS, blk256, 0, stream>>>(h, ln1_w + oD, ln1_b + oD, xln);
    rw_tmix_mix<<<RW_ROWS, blk256, 0, stream>>>(
        xln, maa_x + oD, maa_5 + (size_t)i * 5 * RW_D,
        tm_w1 + (size_t)i * RW_D * 160, tm_w2 + (size_t)i * 160 * RW_D,
        td_w1 + (size_t)i * RW_D * 64,  td_w2 + (size_t)i * 64 * RW_D,
        decay + oD, bxw, bxk, bxv, bxr, bxg, wdec);

    rw_gemm_wmma<<<gemmDD, blk256, 0, stream>>>(bxr, Wr + oDD, rr,
        RW_ROWS, RW_D, RW_D, 0, nullptr, nullptr);
    rw_gemm_wmma<<<gemmDD, blk256, 0, stream>>>(bxk, Wk + oDD, kkb,
        RW_ROWS, RW_D, RW_D, 0, nullptr, nullptr);
    rw_gemm_wmma<<<gemmDD, blk256, 0, stream>>>(bxv, Wv + oDD, vvb,
        RW_ROWS, RW_D, RW_D, 0, nullptr, nullptr);
    rw_gemm_wmma<<<gemmDD, blk256, 0, stream>>>(bxg, Wg + oDD, ggb,
        RW_ROWS, RW_D, RW_D, 1 /*silu*/, nullptr, nullptr);

    rw_wkv<<<RW_B * RW_H, dim3(64), 0, stream>>>(
        rr, kkb, vvb, wdec, faaaa + (size_t)i * RW_H * RW_N, wkvo);

    rw_gn_gate<<<RW_ROWS * RW_H, dim3(64), 0, stream>>>(
        wkvo, lnx_w + oD, lnx_b + oD, ggb, ag);

    // h = h + ag @ Wo   (residual fused)
    rw_gemm_wmma<<<gemmDD, blk256, 0, stream>>>(ag, Wo + oDD, h,
        RW_ROWS, RW_D, RW_D, 0, nullptr, h);

    // ---- cmix ----
    rw_ln<<<RW_ROWS, blk256, 0, stream>>>(h, ln2_w + oD, ln2_b + oD, xln);
    rw_cmix_mix<<<(RW_ROWS * RW_D + 255) / 256, blk256, 0, stream>>>(
        xln, cm_k + oD, cm_r + oD, bxk, bxr);

    // kf = relu(xk @ Wkf)^2
    rw_gemm_wmma<<<gemmDF, blk256, 0, stream>>>(bxk, Wkf + (size_t)i * RW_D * RW_F,
        kf, RW_ROWS, RW_D, RW_F, 2 /*relu^2*/, nullptr, nullptr);
    // rr = sigmoid(xr @ Wrf)
    rw_gemm_wmma<<<gemmDD, blk256, 0, stream>>>(bxr, Wrf + oDD, rr,
        RW_ROWS, RW_D, RW_D, 3 /*sigmoid*/, nullptr, nullptr);
    // h = h + sigmoid(r) * (kf @ Wvf)  (gate + residual fused)
    rw_gemm_wmma<<<gemmDD, blk256, 0, stream>>>(kf, Wvf + (size_t)i * RW_F * RW_D,
        h, RW_ROWS, RW_F, RW_D, 0, rr, h);
  }

  // out = transpose(h) -> [B,D,T]
  rw_transpose_out<<<(RW_B * RW_D * RW_T + 255) / 256, blk256, 0, stream>>>(
      h, (float*)d_out);
}
